// CrossAttention_62208306315918
// MI455X (gfx1250) — compile-verified, matrix operand order checked by reading, not run on hardware
//
#include <hip/hip_runtime.h>
#include <stdint.h>

// CDNA5 (gfx1250) cross-attention: bf16 WMMA GEMMs + f32 softmax.
// Pipeline: proj(Q,K,V) -> scores(+adj) -> softmax -> P@V -> out-proj.
// Tile feeds:
//   MODE_SCORES : TDM tensor_load_to_lds (TENSORcnt), one descriptor per tile
//   MODE_PV/OUT : A via global_load_async_to_lds_b128 (ASYNCcnt)
//   converting paths (f32 weights / V transpose): register-staged fetch/commit

typedef __attribute__((ext_vector_type(16))) __bf16     v16bf;
typedef __attribute__((ext_vector_type(2)))  __bf16     v2bf;
typedef __attribute__((ext_vector_type(8)))  float      v8f;
typedef __attribute__((ext_vector_type(4)))  unsigned   u32x4;
typedef __attribute__((ext_vector_type(8)))  unsigned   u32x8;
typedef __attribute__((ext_vector_type(2)))  unsigned   u32x2;
typedef __attribute__((ext_vector_type(4)))  float      f32x4;

#define BM 128
#define BN 128
#define BK 32
#define LDSW 40   // ushorts per LDS tile row: 32 data + 8 pad (80B stride -> conflict-free b128 frag reads)

enum GemmMode { MODE_PROJ = 0, MODE_SCORES = 1, MODE_PV = 2, MODE_OUT = 3 };

__device__ __forceinline__ unsigned short f32_to_bf16(float f) {
  __bf16 h = (__bf16)f;                     // native cvt path (RNE)
  return __builtin_bit_cast(unsigned short, h);
}

__device__ __forceinline__ unsigned pack_bf16x2(float lo, float hi) {
  v2bf v;
  v.x = (__bf16)lo;
  v.y = (__bf16)hi;                          // folds to v_cvt_pk_bf16_f32
  return __builtin_bit_cast(unsigned, v);
}

// Async 16-byte copy global -> LDS (gfx1250, tracked by ASYNCcnt).
__device__ __forceinline__ void async_copy_b128(void* lds_ptr, const void* gptr) {
  unsigned lds_addr = (unsigned)(uintptr_t)lds_ptr;
  asm volatile("global_load_async_to_lds_b128 %0, %1, off"
               :: "v"(lds_addr), "v"((unsigned long long)(uintptr_t)gptr)
               : "memory");
}
__device__ __forceinline__ void wait_async0() {
  asm volatile("s_wait_asynccnt 0x0" ::: "memory");
}

// TDM: load a 2D tile (tile_d1 rows x tile_d0 bf16 elements) from a row-major
// bf16 tensor (row stride = stride0 elements) into LDS with 16B pad per 64B row,
// reproducing the LDSW=40 padded layout. Descriptor per CDNA5 ISA ch.8.
// Issue from ONE wave; completion via s_wait_tensorcnt + workgroup barrier.
__device__ __forceinline__ void tdm_load_tile_bf16(unsigned lds_addr, const void* gptr,
                                                   unsigned tensor_d0, unsigned tensor_d1,
                                                   unsigned tile_d0, unsigned tile_d1,
                                                   unsigned long long stride0) {
  unsigned long long ga = (unsigned long long)(uintptr_t)gptr;
  u32x4 g0;
  g0[0] = 1u;                                        // count=1, user descriptor
  g0[1] = lds_addr;                                  // LDS byte address
  g0[2] = (unsigned)(ga & 0xFFFFFFFFu);              // global_addr[31:0]
  g0[3] = (unsigned)((ga >> 32) & 0x01FFFFFFu)       // global_addr[56:32]
        | 0x80000000u;                               // type=2 ("image")
  u32x8 g1;
  g1[0] = (1u << 16)                                 // data_size = 2 bytes
        | (1u << 20)                                 // pad_enable
        | (3u << 22)                                 // pad_interval: 16 DWORDs
        | (3u << 25);                                // pad_amount:   4 DWORDs (16B)
  g1[1] = (tensor_d0 & 0xFFFFu) << 16;               // abar_addr=0 | tensor_dim0 lo
  g1[2] = (tensor_d0 >> 16) | ((tensor_d1 & 0xFFFFu) << 16);
  g1[3] = (tensor_d1 >> 16) | (tile_d0 << 16);       // tensor_dim1 hi | tile_dim0
  g1[4] = (tile_d1 & 0xFFFFu);                       // tile_dim1 | tile_dim2=0
  g1[5] = (unsigned)(stride0 & 0xFFFFFFFFu);         // tensor_dim0_stride lo
  g1[6] = (unsigned)((stride0 >> 32) & 0xFFFFu);     // stride hi | dim1_stride=0
  g1[7] = 0u;
  asm volatile("tensor_load_to_lds %0, %1"           // VADDR2/3 = NULL (2D tensor)
               :: "s"(g0), "s"(g1) : "memory");
}

// Load one 16x32 bf16 operand fragment from an LDS tile (row-major, LDSW stride).
// Lane m = lane&15, group g = lane>>4; VGPR0..3: K=g*8.., VGPR4..7: K=16+g*8..
__device__ __forceinline__ v16bf load_frag(const unsigned short* tile, int row, int g) {
  const unsigned short* p = tile + row * LDSW;
  union { u32x4 u[2]; v16bf bf; } fc;
  fc.u[0] = *(const u32x4*)(p + g * 8);
  fc.u[1] = *(const u32x4*)(p + 16 + g * 8);
  return fc.bf;
}

template <int MODE>
__global__ __launch_bounds__(256)
void gemm_wmma_kernel(const void* __restrict__ Av, const void* __restrict__ Bv,
                      void* __restrict__ Cv, const float* __restrict__ aux,
                      int M, int N, int K,
                      long long batchA, long long batchB, long long batchC, long long batchAux,
                      float scale)
{
  constexpr bool USE_TDM   = (MODE == MODE_SCORES);               // A+B pure copies
  constexpr bool USE_ASYNC = (MODE == MODE_PV || MODE == MODE_OUT); // A pure copy

  __shared__ alignas(16) unsigned short sA[2][BM * LDSW];
  __shared__ alignas(16) unsigned short sB[2][BN * LDSW];

  const int tid  = threadIdx.x;
  const int lane = tid & 31;
  const int wave = tid >> 5;
  const int wm   = (wave >> 2) * 64;   // wave row offset inside 128x128 tile
  const int wn   = (wave & 3) * 32;    // wave col offset
  const int ml   = lane & 15;
  const int g    = lane >> 4;

  const int m0 = blockIdx.y * BM;
  const int n0 = blockIdx.x * BN;
  const int b  = blockIdx.z;

  const float*          Af32 = (const float*)Av          + (long long)b * batchA;
  const unsigned short* Abf  = (const unsigned short*)Av + (long long)b * batchA;
  const float*          Bf32 = (const float*)Bv          + (long long)b * batchB;
  const unsigned short* Bbf  = (const unsigned short*)Bv + (long long)b * batchB;

  v8f acc[4][2] = {};

  // Register staging for converting / transposing loaders.
  f32x4 rAf[4];      // MODE_PROJ A (f32)
  f32x4 rBf[4];      // MODE_PROJ / MODE_OUT B (f32 weights)
  u32x4 rBv[2];      // MODE_PV B (bf16 V rows)

  // ---- fetch: issue global loads / async copies / TDM ----
  auto fetch_A = [&](int kt, int buf) {
    if constexpr (MODE == MODE_PROJ) {
      #pragma unroll
      for (int i = 0; i < 4; ++i) {        // 128x32 f32 tile = 1024 float4 chunks
        int c = tid + i * 256;
        int row = c >> 3, c4 = c & 7;
        rAf[i] = *(const f32x4*)(Af32 + (long long)(m0 + row) * K + kt + c4 * 4);
      }
    } else if constexpr (MODE == MODE_SCORES) {
      if (wave == 0)
        tdm_load_tile_bf16((unsigned)(uintptr_t)&sA[buf][0],
                           Abf + (long long)m0 * K + kt,
                           (unsigned)K, (unsigned)M, BK, BM, (unsigned long long)K);
    } else {
      #pragma unroll
      for (int i = 0; i < 2; ++i) {        // 128x32 bf16 tile = 512 b128 chunks
        int c = tid + i * 256;
        int row = c >> 2, c8 = c & 3;
        async_copy_b128(&sA[buf][row * LDSW + c8 * 8],
                        Abf + (long long)(m0 + row) * K + kt + c8 * 8);
      }
    }
  };

  auto fetch_B = [&](int kt, int buf) {
    if constexpr (MODE == MODE_SCORES) {
      if (wave == 0)
        tdm_load_tile_bf16((unsigned)(uintptr_t)&sB[buf][0],
                           Bbf + (long long)n0 * K + kt,
                           (unsigned)K, (unsigned)N, BK, BN, (unsigned long long)K);
    } else if constexpr (MODE == MODE_PV) {
      #pragma unroll
      for (int i = 0; i < 2; ++i) {        // V rows, will transpose on commit
        int c = tid + i * 256;             // 512 chunks: 32 k-rows x 16 chunks of 8
        int k = c >> 4, n8 = c & 15;
        rBv[i] = *(const u32x4*)(Bbf + (long long)(kt + k) * N + n0 + n8 * 8);
      }
    } else {
      #pragma unroll
      for (int i = 0; i < 4; ++i) {        // f32 weights, convert+transpose on commit
        int c = tid + i * 256;             // 1024 chunks: 32 k-rows x 32 float4
        int k = c >> 5, n4 = c & 31;
        rBf[i] = *(const f32x4*)(Bf32 + (long long)(kt + k) * N + n0 + n4 * 4);
      }
    }
  };

  // ---- commit: convert + store staged registers into LDS ----
  auto commit_A = [&](int buf) {
    if constexpr (MODE == MODE_PROJ) {
      #pragma unroll
      for (int i = 0; i < 4; ++i) {
        int c = tid + i * 256;
        int row = c >> 3, c4 = c & 7;
        u32x2 pk;
        pk.x = pack_bf16x2(rAf[i].x, rAf[i].y);
        pk.y = pack_bf16x2(rAf[i].z, rAf[i].w);
        *(u32x2*)(&sA[buf][row * LDSW + c4 * 4]) = pk;
      }
    }
  };

  auto commit_B = [&](int buf) {
    if constexpr (MODE == MODE_PV) {
      #pragma unroll
      for (int i = 0; i < 2; ++i) {
        int c = tid + i * 256;
        int k = c >> 4, n8 = c & 15;
        union { u32x4 v; unsigned short h[8]; } un;
        un.v = rBv[i];
        #pragma unroll
        for (int j = 0; j < 8; ++j)
          sB[buf][(n8 * 8 + j) * LDSW + k] = un.h[j];
      }
    } else if constexpr (MODE == MODE_PROJ || MODE == MODE_OUT) {
      #pragma unroll
      for (int i = 0; i < 4; ++i) {
        int c = tid + i * 256;
        int k = c >> 5, n4 = c & 31;
        sB[buf][(n4 * 4 + 0) * LDSW + k] = f32_to_bf16(rBf[i].x);
        sB[buf][(n4 * 4 + 1) * LDSW + k] = f32_to_bf16(rBf[i].y);
        sB[buf][(n4 * 4 + 2) * LDSW + k] = f32_to_bf16(rBf[i].z);
        sB[buf][(n4 * 4 + 3) * LDSW + k] = f32_to_bf16(rBf[i].w);
      }
    }
  };

  auto drain = [&]() {
    if constexpr (USE_ASYNC) wait_async0();
    if constexpr (USE_TDM) {
      if (wave == 0) __builtin_amdgcn_s_wait_tensorcnt((short)0);
    }
  };

  // Prologue: fill buffer 0.
  fetch_A(0, 0);
  fetch_B(0, 0);
  commit_A(0);
  commit_B(0);
  drain();
  __syncthreads();

  const int nk = K / BK;
  for (int t = 0; t < nk; ++t) {
    const int buf = t & 1;
    const bool pf = (t + 1 < nk);
    if (pf) {                              // issue next tile's loads before compute
      fetch_A((t + 1) * BK, buf ^ 1);
      fetch_B((t + 1) * BK, buf ^ 1);
    }

    v16bf afr[4], bfr[2];
    #pragma unroll
    for (int i = 0; i < 4; ++i) afr[i] = load_frag(sA[buf], wm + i * 16 + ml, g);
    #pragma unroll
    for (int j = 0; j < 2; ++j) bfr[j] = load_frag(sB[buf], wn + j * 16 + ml, g);

    #pragma unroll
    for (int i = 0; i < 4; ++i)
      #pragma unroll
      for (int j = 0; j < 2; ++j)
        acc[i][j] = __builtin_amdgcn_wmma_f32_16x16x32_bf16(
            false, afr[i], false, bfr[j], (short)0, acc[i][j], false, false);

    if (pf) {                              // convert/store after compute hid the latency
      commit_A(buf ^ 1);
      commit_B(buf ^ 1);
      drain();
    }
    __syncthreads();
  }

  // Epilogue. C fragment layout: lane n = ml, VGPR r -> row 8*g + r.
  if constexpr (MODE == MODE_PROJ || MODE == MODE_PV) {
    unsigned short* C = (unsigned short*)Cv + (long long)b * batchC;
    #pragma unroll
    for (int i = 0; i < 4; ++i) {
      #pragma unroll
      for (int j = 0; j < 2; ++j) {
        const int col = n0 + wn + j * 16 + ml;
        float bias = 0.f;
        if constexpr (MODE == MODE_PROJ) bias = aux[col];
        #pragma unroll
        for (int r = 0; r < 8; ++r) {
          const int row = m0 + wm + i * 16 + g * 8 + r;
          C[(long long)row * N + col] = f32_to_bf16(acc[i][j][r] + bias);
        }
      }
    }
  } else if constexpr (MODE == MODE_SCORES) {
    float* C = (float*)Cv + (long long)b * batchC;
    const float* adj = aux + (long long)b * batchAux;
    #pragma unroll
    for (int i = 0; i < 4; ++i) {
      #pragma unroll
      for (int j = 0; j < 2; ++j) {
        const int col = n0 + wn + j * 16 + ml;
        #pragma unroll
        for (int r = 0; r < 8; ++r) {
          const int row = m0 + wm + i * 16 + g * 8 + r;
          C[(long long)row * N + col] =
              acc[i][j][r] * scale + adj[(long long)row * N + col];
        }
      }
    }
  } else {  // MODE_OUT
    float* C = (float*)Cv + (long long)b * batchC;
    #pragma unroll
    for (int i = 0; i < 4; ++i) {
      #pragma unroll
      for (int j = 0; j < 2; ++j) {
        const int col = n0 + wn + j * 16 + ml;
        const float bias = aux[col];
        #pragma unroll
        for (int r = 0; r < 8; ++r) {
          const int row = m0 + wm + i * 16 + g * 8 + r;
          C[(long long)row * N + col] = acc[i][j][r] + bias;
        }
      }
    }
  }
}

// Row softmax over T=2048 f32 logits, writes bf16 probabilities. One block per row.
__global__ __launch_bounds__(256)
void softmax_kernel(const float* __restrict__ att, unsigned short* __restrict__ P, int T)
{
  const long long row = blockIdx.x;
  const float* x = att + row * (long long)T;
  __shared__ float red[8];
  const int tid = threadIdx.x;

  float v[8];
  float mx = -3.4e38f;
  #pragma unroll
  for (int i = 0; i < 8; ++i) { v[i] = x[tid + i * 256]; mx = fmaxf(mx, v[i]); }
  #pragma unroll
  for (int o = 16; o > 0; o >>= 1) mx = fmaxf(mx, __shfl_xor(mx, o, 32));
  if ((tid & 31) == 0) red[tid >> 5] = mx;
  __syncthreads();
  if (tid == 0) {
    float m = red[0];
    #pragma unroll
    for (int i = 1; i < 8; ++i) m = fmaxf(m, red[i]);
    red[0] = m;
  }
  __syncthreads();
  mx = red[0];
  __syncthreads();

  float s = 0.f;
  #pragma unroll
  for (int i = 0; i < 8; ++i) { v[i] = __expf(v[i] - mx); s += v[i]; }
  #pragma unroll
  for (int o = 16; o > 0; o >>= 1) s += __shfl_xor(s, o, 32);
  if ((tid & 31) == 0) red[tid >> 5] = s;
  __syncthreads();
  if (tid == 0) {
    float t = 0.f;
    #pragma unroll
    for (int i = 0; i < 8; ++i) t += red[i];
    red[0] = t;
  }
  __syncthreads();
  const float rs = 1.0f / red[0];

  unsigned short* p = P + row * (long long)T;
  #pragma unroll
  for (int i = 0; i < 8; ++i) p[tid + i * 256] = f32_to_bf16(v[i] * rs);
}

extern "C" void kernel_launch(void* const* d_in, const int* in_sizes, int n_in,
                              void* d_out, int out_size, void* d_ws, size_t ws_size,
                              hipStream_t stream)
{
  (void)in_sizes; (void)n_in; (void)out_size; (void)ws_size;

  const float* x   = (const float*)d_in[0];
  const float* y   = (const float*)d_in[1];
  const float* adj = (const float*)d_in[2];
  const float* Wq  = (const float*)d_in[3];
  const float* bq  = (const float*)d_in[4];
  const float* Wk  = (const float*)d_in[5];
  const float* bk  = (const float*)d_in[6];
  const float* Wv  = (const float*)d_in[7];
  const float* bv  = (const float*)d_in[8];
  const float* Wo  = (const float*)d_in[9];
  const float* bo  = (const float*)d_in[10];
  float* out = (float*)d_out;

  const int D = 1024, S = 2048, Bn = 4;
  const int MS = Bn * S;                      // 8192 flattened rows
  const long long MB = 1024ll * 1024ll;

  char* ws = (char*)d_ws;
  unsigned short* Qbf = (unsigned short*)(ws + 0);          // 16 MB
  unsigned short* Kbf = (unsigned short*)(ws + 16 * MB);    // 16 MB
  unsigned short* Vbf = (unsigned short*)(ws + 32 * MB);    // 16 MB
  float*          att = (float*)(ws + 48 * MB);             // 64 MB
  unsigned short* P   = (unsigned short*)(ws + 112 * MB);   // 32 MB
  unsigned short* tmp = (unsigned short*)(ws + 144 * MB);   // 16 MB (total 160 MB)

  dim3 blk(256);

  // Q/K/V projections: [8192,1024] = [8192,1024]f32 @ [1024,1024]f32 + bias -> bf16
  gemm_wmma_kernel<MODE_PROJ><<<dim3(D / BN, MS / BM, 1), blk, 0, stream>>>(
      x, Wq, Qbf, bq, MS, D, D, 0, 0, 0, 0, 1.f);
  gemm_wmma_kernel<MODE_PROJ><<<dim3(D / BN, MS / BM, 1), blk, 0, stream>>>(
      y, Wk, Kbf, bk, MS, D, D, 0, 0, 0, 0, 1.f);
  gemm_wmma_kernel<MODE_PROJ><<<dim3(D / BN, MS / BM, 1), blk, 0, stream>>>(
      y, Wv, Vbf, bv, MS, D, D, 0, 0, 0, 0, 1.f);

  // att[b] = Q[b] @ K[b]^T / sqrt(D) + adj[b]   (f32 logits, TDM-fed tiles)
  gemm_wmma_kernel<MODE_SCORES><<<dim3(S / BN, S / BM, Bn), blk, 0, stream>>>(
      Qbf, Kbf, att, adj, S, S, D,
      (long long)S * D, (long long)S * D, (long long)S * S, (long long)S * S,
      0.03125f /* 1/sqrt(1024) */);

  // row softmax -> bf16 probabilities
  softmax_kernel<<<dim3((unsigned)MS), blk, 0, stream>>>(att, P, S);

  // tmp[b] = P[b] @ V[b]   (bf16)
  gemm_wmma_kernel<MODE_PV><<<dim3(D / BN, S / BM, Bn), blk, 0, stream>>>(
      P, Vbf, tmp, nullptr, S, D, S,
      (long long)S * S, (long long)S * D, (long long)S * D, 0, 1.f);

  // out = tmp @ Wo + bo   (f32)
  gemm_wmma_kernel<MODE_OUT><<<dim3(D / BN, MS / BM, 1), blk, 0, stream>>>(
      tmp, Wo, out, bo, MS, D, D, 0, 0, 0, 0, 1.f);
}
